// GCN_21028159881585
// MI455X (gfx1250) — compile-verified, matrix operand order checked by reading, not run on hardware
//
#include <hip/hip_runtime.h>
#include <stdint.h>

// ---------------------------------------------------------------------------
// Types for CDNA5 WMMA (wave32): v_wmma_f32_16x16x32_bf16
// ---------------------------------------------------------------------------
typedef __attribute__((ext_vector_type(16))) __bf16 v16bf;
typedef __attribute__((ext_vector_type(8)))  float  v8f;

#define DEVFN static __device__ __forceinline__

// Per the CDNA5 ISA 16-bit A-matrix layout (16x32, MxK):
//   lane L holds row m = L%16; khalf = L/16.
//   VGPR0..3  : K = khalf*8 + {0..7}        (consecutive in memory)
//   VGPR4..7  : K = 16 + khalf*8 + {0..7}   (consecutive in memory)
// So a fragment is exactly two 16-byte contiguous runs -> two b128 loads.
// B (K x N) is stored transposed (N x Kp) so its fragment has the same shape.
DEVFN v16bf load_frag16(const __bf16* p0, const __bf16* p1) {
  union { uint32_t u[8]; v16bf v; } r;
  const uint4 a = *reinterpret_cast<const uint4*>(p0);
  const uint4 b = *reinterpret_cast<const uint4*>(p1);
  r.u[0] = a.x; r.u[1] = a.y; r.u[2] = a.z; r.u[3] = a.w;
  r.u[4] = b.x; r.u[5] = b.y; r.u[6] = b.z; r.u[7] = b.w;
  return r.v;
}

// ---------------------------------------------------------------------------
// WMMA GEMM:  C[M,N] = A[M,Kp](bf16) * Bt[N,Kp]^T(bf16)
// Epilogue: optional per-row scale (dinv), per-col bias, relu; f32 or bf16 out.
// Block = 256 threads = 8 waves; wave w computes rows [tm*16,tm*16+16) and a
// 64-wide column strip: four 16x16 tiles sharing one A fragment (NT=4).
// The K loop is software-pipelined (register double-buffered): chunk kc+32's
// ten b128 loads are issued BEFORE the four WMMAs of chunk kc, forcing the
// next fragments live across the WMMAs so waits become partial (latency of a
// full chunk is hidden inside a single wave, on top of 8-wave occupancy).
// Requires: M % 16 == 0, N % 64 == 0, Kp % 32 == 0 and Kp >= 32 (zero-padded).
// ---------------------------------------------------------------------------
__global__ __launch_bounds__(256) void gcn_gemm_bf16_wmma(
    const __bf16* __restrict__ A, const __bf16* __restrict__ Bt,
    float* __restrict__ Cf, __bf16* __restrict__ Cb,
    const float* __restrict__ rowscale, const float* __restrict__ bias,
    int M, int N, int Kp, int relu)
{
  const int lane = threadIdx.x & 31;
  const int wave = threadIdx.x >> 5;
  const int tm   = blockIdx.x * 8 + wave;      // 16-row tile index
  if (tm * 16 >= M) return;
  const int tn0  = blockIdx.y * 4;             // first of four 16-col tiles
  const int half = lane >> 4;                  // 0..1 (K-half select)
  const int lid  = lane & 15;                  // row (A) / col (B) within tile
  const int kh   = half * 8;

  const __bf16* arow  = A + (size_t)(tm * 16 + lid) * Kp + kh;
  const __bf16* brow0 = Bt + (size_t)((tn0 + 0) * 16 + lid) * Kp + kh;
  const __bf16* brow1 = Bt + (size_t)((tn0 + 1) * 16 + lid) * Kp + kh;
  const __bf16* brow2 = Bt + (size_t)((tn0 + 2) * 16 + lid) * Kp + kh;
  const __bf16* brow3 = Bt + (size_t)((tn0 + 3) * 16 + lid) * Kp + kh;

  v8f acc0 = {0.f,0.f,0.f,0.f,0.f,0.f,0.f,0.f};
  v8f acc1 = {0.f,0.f,0.f,0.f,0.f,0.f,0.f,0.f};
  v8f acc2 = {0.f,0.f,0.f,0.f,0.f,0.f,0.f,0.f};
  v8f acc3 = {0.f,0.f,0.f,0.f,0.f,0.f,0.f,0.f};

  // Prologue: preload chunk 0.
  v16bf af  = load_frag16(arow,  arow  + 16);
  v16bf bf0 = load_frag16(brow0, brow0 + 16);
  v16bf bf1 = load_frag16(brow1, brow1 + 16);
  v16bf bf2 = load_frag16(brow2, brow2 + 16);
  v16bf bf3 = load_frag16(brow3, brow3 + 16);

  for (int kc = 32; kc < Kp; kc += 32) {
    // Issue next chunk's loads first (distinct SSA values -> must stay live
    // across the WMMAs below -> distinct registers, partial waits).
    const v16bf afn  = load_frag16(arow  + kc, arow  + kc + 16);
    const v16bf bf0n = load_frag16(brow0 + kc, brow0 + kc + 16);
    const v16bf bf1n = load_frag16(brow1 + kc, brow1 + kc + 16);
    const v16bf bf2n = load_frag16(brow2 + kc, brow2 + kc + 16);
    const v16bf bf3n = load_frag16(brow3 + kc, brow3 + kc + 16);

    acc0 = __builtin_amdgcn_wmma_f32_16x16x32_bf16(
        false, af, false, bf0, (short)0, acc0, false, false);
    acc1 = __builtin_amdgcn_wmma_f32_16x16x32_bf16(
        false, af, false, bf1, (short)0, acc1, false, false);
    acc2 = __builtin_amdgcn_wmma_f32_16x16x32_bf16(
        false, af, false, bf2, (short)0, acc2, false, false);
    acc3 = __builtin_amdgcn_wmma_f32_16x16x32_bf16(
        false, af, false, bf3, (short)0, acc3, false, false);

    af = afn; bf0 = bf0n; bf1 = bf1n; bf2 = bf2n; bf3 = bf3n;
  }

  // Epilogue: last chunk.
  acc0 = __builtin_amdgcn_wmma_f32_16x16x32_bf16(
      false, af, false, bf0, (short)0, acc0, false, false);
  acc1 = __builtin_amdgcn_wmma_f32_16x16x32_bf16(
      false, af, false, bf1, (short)0, acc1, false, false);
  acc2 = __builtin_amdgcn_wmma_f32_16x16x32_bf16(
      false, af, false, bf2, (short)0, acc2, false, false);
  acc3 = __builtin_amdgcn_wmma_f32_16x16x32_bf16(
      false, af, false, bf3, (short)0, acc3, false, false);

  // C/D layout: VGPR r -> row m = tm*16 + 8*half + r; col n = tile*16 + lid
  v8f accs[4] = {acc0, acc1, acc2, acc3};
  #pragma unroll
  for (int j = 0; j < 4; ++j) {
    const int   n  = (tn0 + j) * 16 + lid;
    const float bv = bias ? bias[n] : 0.0f;
    #pragma unroll
    for (int r = 0; r < 8; ++r) {
      const int   mr = tm * 16 + kh + r;
      const float sc = rowscale ? rowscale[mr] : 1.0f;
      float v = accs[j][r] * sc + bv;
      if (relu) v = fmaxf(v, 0.0f);
      const size_t o = (size_t)mr * N + n;
      if (Cf) Cf[o] = v;
      else    Cb[o] = (__bf16)v;
    }
  }
}

// ---------------------------------------------------------------------------
// Degree accumulation (self-loop folded in as +1 inside rsqrt)
// ---------------------------------------------------------------------------
__global__ void deg_kernel(const int* __restrict__ dst,
                           float* __restrict__ deg, int E) {
  int e = blockIdx.x * blockDim.x + threadIdx.x;
  if (e < E) atomicAdd(&deg[dst[e]], 1.0f);
}

__global__ void dinv_kernel(float* __restrict__ deg, int Nn) {
  int i = blockIdx.x * blockDim.x + threadIdx.x;
  if (i < Nn) deg[i] = rsqrtf(deg[i] + 1.0f);   // in-place deg -> dinv
}

// ---------------------------------------------------------------------------
// f32 -> bf16 with zero K-padding (writes packed pairs as u32)
// ---------------------------------------------------------------------------
__global__ void cvt_pad_kernel(const float* __restrict__ X,
                               __bf16* __restrict__ Xb,
                               int M, int K, int Kp) {
  size_t t = (size_t)blockIdx.x * blockDim.x + threadIdx.x;
  size_t total = (size_t)M * (size_t)(Kp >> 1);
  if (t >= total) return;
  int m  = (int)(t / (size_t)(Kp >> 1));
  int kp = (int)(t % (size_t)(Kp >> 1)) * 2;
  float f0 = (kp     < K) ? X[(size_t)m * K + kp]     : 0.0f;
  float f1 = (kp + 1 < K) ? X[(size_t)m * K + kp + 1] : 0.0f;
  union { __bf16 h[2]; uint32_t u; } p;
  p.h[0] = (__bf16)f0; p.h[1] = (__bf16)f1;
  *reinterpret_cast<uint32_t*>(Xb + (size_t)m * Kp + kp) = p.u;
}

// W[K,N] (row-major f32) -> Wt[N,Kp] (bf16, zero-padded K)
__global__ void wtrans_kernel(const float* __restrict__ W,
                              __bf16* __restrict__ Wt,
                              int K, int N, int Kp) {
  int idx = blockIdx.x * blockDim.x + threadIdx.x;
  if (idx >= N * Kp) return;
  int n = idx / Kp, k = idx % Kp;
  float v = (k < K) ? W[(size_t)k * N + n] : 0.0f;
  Wt[idx] = (__bf16)v;
}

// ---------------------------------------------------------------------------
// Edge scatter: S[dst] += P[src]   (float4 gather + 4x f32 atomics)
// ---------------------------------------------------------------------------
__global__ void scatter_kernel(const int* __restrict__ src,
                               const int* __restrict__ dst,
                               const float* __restrict__ P,
                               float* __restrict__ S,
                               int E, int Hc /* = H/4 */, int H) {
  size_t t = (size_t)blockIdx.x * blockDim.x + threadIdx.x;
  size_t total = (size_t)E * (size_t)Hc;
  if (t >= total) return;
  int e = (int)(t / (size_t)Hc);
  int c = (int)(t % (size_t)Hc) * 4;
  int s = src[e], d = dst[e];
  const float4 v = *reinterpret_cast<const float4*>(P + (size_t)s * H + c);
  float* o = S + (size_t)d * H + c;
  atomicAdd(o + 0, v.x); atomicAdd(o + 1, v.y);
  atomicAdd(o + 2, v.z); atomicAdd(o + 3, v.w);
}

// h = relu(dinv * (S + P) + b)  (P term = self loop), output bf16 for next GEMM
__global__ void post_kernel(const float* __restrict__ S,
                            const float* __restrict__ P,
                            const float* __restrict__ dinv,
                            const float* __restrict__ bias,
                            __bf16* __restrict__ Hout, int Nn, int H) {
  size_t t = (size_t)blockIdx.x * blockDim.x + threadIdx.x;
  if (t >= (size_t)Nn * H) return;
  int i = (int)(t / (size_t)H), c = (int)(t % (size_t)H);
  float v = dinv[i] * (S[t] + P[t]) + bias[c];
  Hout[t] = (__bf16)fmaxf(v, 0.0f);
}

// z[b, :] = concat(h2[idx[b]], wt[b], mut[b], zero-pad)  -> bf16 [B, Zp]
__global__ void buildz_kernel(const __bf16* __restrict__ h2,
                              const int* __restrict__ idx,
                              const float* __restrict__ wt,
                              const float* __restrict__ mut,
                              __bf16* __restrict__ z,
                              int B, int H, int Z, int Zp) {
  size_t t = (size_t)blockIdx.x * blockDim.x + threadIdx.x;
  if (t >= (size_t)B * Zp) return;
  int b = (int)(t / (size_t)Zp), c = (int)(t % (size_t)Zp);
  __bf16 v;
  if (c < H)            v = h2[(size_t)idx[b] * H + c];
  else if (c < H + 20)  v = (__bf16)wt[(size_t)b * 20 + (c - H)];
  else if (c < Z)       v = (__bf16)mut[(size_t)b * 20 + (c - H - 20)];
  else                  v = (__bf16)0.0f;
  z[t] = v;
}

// out[b] = z2[b,:] . Wh3 + bh3   (K = 64, trivial)
__global__ void final_kernel(const __bf16* __restrict__ z2,
                             const float* __restrict__ Wh3,
                             const float* __restrict__ bh3,
                             float* __restrict__ out, int B, int K) {
  int b = blockIdx.x * blockDim.x + threadIdx.x;
  if (b >= B) return;
  float acc = 0.0f;
  const __bf16* row = z2 + (size_t)b * K;
  for (int k = 0; k < K; ++k) acc += (float)row[k] * Wh3[k];
  out[b] = acc + bh3[0];
}

// ---------------------------------------------------------------------------
// Host-side orchestration
// ---------------------------------------------------------------------------
static inline unsigned cdiv(size_t a, size_t b) { return (unsigned)((a + b - 1) / b); }
static inline size_t   al256(size_t x)          { return (x + 255) & ~(size_t)255; }

extern "C" void kernel_launch(void* const* d_in, const int* in_sizes, int n_in,
                              void* d_out, int out_size, void* d_ws, size_t ws_size,
                              hipStream_t stream) {
  // ---- inputs (setup_inputs order) ----
  const float* x      = (const float*)d_in[0];
  const float* wt     = (const float*)d_in[1];
  const float* mut    = (const float*)d_in[2];
  const float* Wc1    = (const float*)d_in[3];
  const float* bc1    = (const float*)d_in[4];
  const float* Wc2    = (const float*)d_in[5];
  const float* bc2    = (const float*)d_in[6];
  const float* Wh1    = (const float*)d_in[7];
  const float* bh1    = (const float*)d_in[8];
  const float* Wh2    = (const float*)d_in[9];
  const float* bh2    = (const float*)d_in[10];
  const float* Wh3    = (const float*)d_in[11];
  const float* bh3    = (const float*)d_in[12];
  const int*   srcp   = (const int*)d_in[13];              // edge_index[0]
  const int*   vidx   = (const int*)d_in[14];
  float*       outp   = (float*)d_out;

  // ---- dims (derived where possible) ----
  const int H    = in_sizes[4];                 // 256
  const int HH   = in_sizes[8];                 // 128
  const int HQ   = in_sizes[10];                // 64
  const int DIN  = in_sizes[3] / H;             // 1281
  const int Nn   = in_sizes[0] / DIN;           // 50000
  const int E    = in_sizes[13] / 2;            // 800000
  const int Bb   = in_sizes[14];                // 4096
  const int DINP = (DIN + 31) & ~31;            // 1312
  const int Z    = H + 40;                      // 296
  const int ZP   = (Z + 31) & ~31;              // 320
  const int* dstp = srcp + E;                   // edge_index[1]

  // ---- workspace carve-out ----
  char* ws = (char*)d_ws;
  size_t off = 0;
  __bf16* Xbf  = (__bf16*)(ws + off); off = al256(off + (size_t)Nn * DINP * 2);
  __bf16* Wc1t = (__bf16*)(ws + off); off = al256(off + (size_t)H  * DINP * 2);
  __bf16* Wc2t = (__bf16*)(ws + off); off = al256(off + (size_t)H  * H    * 2);
  __bf16* Wh1t = (__bf16*)(ws + off); off = al256(off + (size_t)HH * ZP   * 2);
  __bf16* Wh2t = (__bf16*)(ws + off); off = al256(off + (size_t)HQ * HH   * 2);
  float*  dinv = (float*) (ws + off); off = al256(off + (size_t)Nn * 4);
  float*  P    = (float*) (ws + off); off = al256(off + (size_t)Nn * H * 4);
  float*  S    = (float*) (ws + off); off = al256(off + (size_t)Nn * H * 4);
  __bf16* hbf  = (__bf16*)(ws + off); off = al256(off + (size_t)Nn * H * 2);
  __bf16* zbf  = (__bf16*)(ws + off); off = al256(off + (size_t)Bb * ZP * 2);
  __bf16* z1bf = (__bf16*)(ws + off); off = al256(off + (size_t)Bb * HH * 2);
  __bf16* z2bf = (__bf16*)(ws + off); off = al256(off + (size_t)Bb * HQ * 2);
  (void)ws_size; (void)n_in; (void)out_size;

  const int T = 256;

  // 1) degrees -> dinv (in place)
  hipMemsetAsync(dinv, 0, (size_t)Nn * 4, stream);
  deg_kernel <<<cdiv(E, T), T, 0, stream>>>(dstp, dinv, E);
  dinv_kernel<<<cdiv(Nn, T), T, 0, stream>>>(dinv, Nn);

  // 2) convert/pad activations + weights to bf16
  cvt_pad_kernel<<<cdiv((size_t)Nn * (DINP / 2), T), T, 0, stream>>>(x, Xbf, Nn, DIN, DINP);
  wtrans_kernel <<<cdiv((size_t)H  * DINP, T), T, 0, stream>>>(Wc1, Wc1t, DIN, H, DINP);
  wtrans_kernel <<<cdiv((size_t)H  * H,    T), T, 0, stream>>>(Wc2, Wc2t, H,   H, H);
  wtrans_kernel <<<cdiv((size_t)HH * ZP,   T), T, 0, stream>>>(Wh1, Wh1t, Z,  HH, ZP);
  wtrans_kernel <<<cdiv((size_t)HQ * HH,   T), T, 0, stream>>>(Wh2, Wh2t, HH, HQ, HH);

  // 3) layer 1: P = dinv * (X @ Wc1); S = scatter(P); h1 = relu(dinv*(S+P)+b)
  dim3 gNode(cdiv((size_t)Nn / 16, 8), H / 64);
  gcn_gemm_bf16_wmma<<<gNode, T, 0, stream>>>(Xbf, Wc1t, P, nullptr, dinv, nullptr,
                                              Nn, H, DINP, 0);
  hipMemsetAsync(S, 0, (size_t)Nn * H * 4, stream);
  scatter_kernel<<<cdiv((size_t)E * (H / 4), T), T, 0, stream>>>(srcp, dstp, P, S, E, H / 4, H);
  post_kernel<<<cdiv((size_t)Nn * H, T), T, 0, stream>>>(S, P, dinv, bc1, hbf, Nn, H);

  // 4) layer 2 (same shape, Kp = H)
  gcn_gemm_bf16_wmma<<<gNode, T, 0, stream>>>(hbf, Wc2t, P, nullptr, dinv, nullptr,
                                              Nn, H, H, 0);
  hipMemsetAsync(S, 0, (size_t)Nn * H * 4, stream);
  scatter_kernel<<<cdiv((size_t)E * (H / 4), T), T, 0, stream>>>(srcp, dstp, P, S, E, H / 4, H);
  post_kernel<<<cdiv((size_t)Nn * H, T), T, 0, stream>>>(S, P, dinv, bc2, hbf, Nn, H);

  // 5) MLP head
  buildz_kernel<<<cdiv((size_t)Bb * ZP, T), T, 0, stream>>>(hbf, vidx, wt, mut, zbf, Bb, H, Z, ZP);
  dim3 gM1(cdiv((size_t)Bb / 16, 8), HH / 64);
  gcn_gemm_bf16_wmma<<<gM1, T, 0, stream>>>(zbf, Wh1t, nullptr, z1bf, nullptr, bh1,
                                            Bb, HH, ZP, 1);
  dim3 gM2(cdiv((size_t)Bb / 16, 8), HQ / 64);
  gcn_gemm_bf16_wmma<<<gM2, T, 0, stream>>>(z1bf, Wh2t, nullptr, z2bf, nullptr, bh2,
                                            Bb, HQ, HH, 1);
  final_kernel<<<cdiv(Bb, T), T, 0, stream>>>(z2bf, Wh3, bh3, outp, Bb, HQ);
}
